// GroupedQueryAttention_43834436223189
// MI455X (gfx1250) — matrix operand order repaired and checked by reading, not compiled, not run on hardware
//
#include <hip/hip_runtime.h>
#include <hip/hip_bf16.h>

// ---------------------------------------------------------------------------
// GQA attention (B=2, L=2048, D=2048, H=16, KVH=4, HD=128) for gfx1250.
// All matmuls on v_wmma_f32_16x16x32_bf16 (f32 accumulation); flash-attention
// with K/V tiles double-buffered into LDS via global_load_async_to_lds_b128
// (ASYNCcnt-tracked) so global latency overlaps the WMMA pipe.
// ---------------------------------------------------------------------------

typedef __bf16 bf16_t;
typedef __attribute__((ext_vector_type(16))) __bf16 v16bf;
typedef __attribute__((ext_vector_type(8)))  __bf16 v8bf;
typedef __attribute__((ext_vector_type(8)))  float  v8f;

#define WMMA_BF16(a, b, c) \
    __builtin_amdgcn_wmma_f32_16x16x32_bf16(false, (a), false, (b), (short)0, (c), false, false)

__constant__ constexpr int cB   = 2;
__constant__ constexpr int cL   = 2048;
__constant__ constexpr int cD   = 2048;
__constant__ constexpr int cH   = 16;
__constant__ constexpr int cKVH = 4;
__constant__ constexpr int cHD  = 128;
__constant__ constexpr int cG   = cH / cKVH;       // 4
__constant__ constexpr int cKVD = cKVH * cHD;      // 512
__constant__ constexpr float cTHETA = 10000.0f;

// --- CDNA5 async copy: 16B per lane, global -> LDS, tracked by ASYNCcnt ----
__device__ __forceinline__ void async_copy16(unsigned lds_off, const void* gptr) {
    asm volatile("global_load_async_to_lds_b128 %0, %1, off"
                 :: "v"(lds_off), "v"(gptr) : "memory");
}
__device__ __forceinline__ void wait_async_0()  {
    asm volatile("s_wait_asynccnt 0x0" ::: "memory");
}
__device__ __forceinline__ void wait_async_32() {
    asm volatile("s_wait_asynccnt 0x20" ::: "memory");
}

// ---------------------------------------------------------------------------
// elementwise f32 -> bf16 cast
// ---------------------------------------------------------------------------
__global__ void cast_f32_bf16(const float* __restrict__ in, bf16_t* __restrict__ out, size_t n) {
    size_t i = (size_t)blockIdx.x * blockDim.x + threadIdx.x;
    size_t stride = (size_t)gridDim.x * blockDim.x;
    for (; i < n; i += stride) out[i] = (bf16_t)in[i];
}

// ---------------------------------------------------------------------------
// Y[M,N] = X[M,K] @ W[N,K]^T   (torch Linear), bf16 inputs, f32 accumulate.
// Block = 128 threads = 4 waves; each wave computes a 16x64 output strip
// (4 WMMA tiles sharing one A load). Grid: (M/16, N/256).
// A-operand: lane m holds k = 0..15 (lo half) / 16..31 (hi), contiguous.
// B-operand: lane n holds 16 contiguous k from W row n (x @ W.T => direct).
// ---------------------------------------------------------------------------
template <bool F32OUT>
__global__ __launch_bounds__(128) void gemm_xwT(const bf16_t* __restrict__ X,
                                                const bf16_t* __restrict__ W,
                                                float* __restrict__ Yf,
                                                bf16_t* __restrict__ Yb,
                                                int M, int N, int K) {
    const int lane = threadIdx.x & 31;
    const int wave = threadIdx.x >> 5;
    const int nn = lane & 15;
    const int hi = lane >> 4;
    const int m0 = blockIdx.x * 16;
    const int n0 = blockIdx.y * 256 + wave * 64;

    const bf16_t* __restrict__ Xrow = X + (size_t)(m0 + nn) * K;

    v8f acc[4] = {};
    for (int k0 = 0; k0 < K; k0 += 32) {
        v8bf a0 = *(const v8bf*)(Xrow + k0 + hi * 16);
        v8bf a1 = *(const v8bf*)(Xrow + k0 + hi * 16 + 8);
        v16bf a;
#pragma unroll
        for (int i = 0; i < 8; ++i) { a[i] = a0[i]; a[8 + i] = a1[i]; }
#pragma unroll
        for (int t = 0; t < 4; ++t) {
            const bf16_t* wp = W + (size_t)(n0 + t * 16 + nn) * K + k0 + hi * 16;
            v16bf b = *(const v16bf*)wp;
            acc[t] = WMMA_BF16(a, b, acc[t]);
        }
    }
    // C/D layout: row = m0 + c + 8*hi, col = nn
#pragma unroll
    for (int t = 0; t < 4; ++t) {
#pragma unroll
        for (int c = 0; c < 8; ++c) {
            size_t idx = (size_t)(m0 + c + 8 * hi) * N + (n0 + t * 16 + nn);
            if (F32OUT) Yf[idx] = acc[t][c];
            else        Yb[idx] = (bf16_t)acc[t][c];
        }
    }
}

// ---------------------------------------------------------------------------
// RoPE in place on Q [B,L,H*HD] and K [B,L,KVH*HD] (bf16), rotate-half form.
// ---------------------------------------------------------------------------
__global__ __launch_bounds__(256) void rope_kernel(bf16_t* __restrict__ q, bf16_t* __restrict__ k) {
    const int l = blockIdx.x;
    const int b = blockIdx.y;
    const float pos = (float)l;
    for (int t = threadIdx.x; t < (cH + cKVH) * (cHD / 2); t += 256) {
        const int head = t / (cHD / 2);
        const int i    = t % (cHD / 2);
        const float freq = __powf(cTHETA, -2.0f * (float)i / (float)cHD);
        float s, c;
        __sincosf(pos * freq, &s, &c);
        bf16_t* p;
        if (head < cH) p = q + (size_t)(b * cL + l) * cD   + head * cHD;
        else           p = k + (size_t)(b * cL + l) * cKVD + (head - cH) * cHD;
        const float x1 = (float)p[i];
        const float x2 = (float)p[i + cHD / 2];
        p[i]            = (bf16_t)(x1 * c - x2 * s);
        p[i + cHD / 2]  = (bf16_t)(x2 * c + x1 * s);
    }
}

// ---------------------------------------------------------------------------
// V [B,L,KVH*HD] -> Vt [B,KVH,HD,L]  (so PV B-operand reads contiguous keys)
// ---------------------------------------------------------------------------
__global__ void transpose_v(const bf16_t* __restrict__ V, bf16_t* __restrict__ Vt) {
    const size_t n = (size_t)cB * cKVH * cHD * cL;
    size_t i = (size_t)blockIdx.x * blockDim.x + threadIdx.x;
    size_t stride = (size_t)gridDim.x * blockDim.x;
    for (; i < n; i += stride) {
        const int l  = (int)(i & (cL - 1));
        size_t r = i >> 11;
        const int d  = (int)(r & (cHD - 1));  r >>= 7;
        const int kh = (int)(r & (cKVH - 1)); r >>= 2;
        const int b  = (int)r;
        Vt[i] = V[(size_t)(b * cL + l) * cKVD + kh * cHD + d];
    }
}

// ---------------------------------------------------------------------------
// Flash attention: one wave = one 16-query tile of one (b, h).
// Causal loop over 32-key blocks. K (32x128) and V^T (128x32) tiles are
// double-buffered in LDS via global_load_async_to_lds_b128; while WMMAs
// consume block j, the async engine fills block j+1 (s_wait_asynccnt 32).
// S = Q K^T (8 WMMAs), online softmax with half-wave shfl_xor reductions,
// P restaged via LDS into A-operand layout, O += P V (8 WMMAs, 16x128 f32).
// ---------------------------------------------------------------------------
__global__ __launch_bounds__(32) void gqa_flash_attn(const bf16_t* __restrict__ Q,
                                                     const bf16_t* __restrict__ K,
                                                     const bf16_t* __restrict__ Vt,
                                                     bf16_t* __restrict__ O) {
    const int q0 = blockIdx.x * 16;
    const int h  = blockIdx.y;
    const int b  = blockIdx.z;
    const int kh = h / cG;
    const int lane = threadIdx.x;
    const int nn = lane & 15;
    const int hi = lane >> 4;
    const float scale = 0.08838834764831845f;   // 1/sqrt(128)

    __shared__ alignas(32) bf16_t smemK[2][32 * 128];   // [buf][j][d]  8KB each
    __shared__ alignas(32) bf16_t smemV[2][128 * 32];   // [buf][d][j]  8KB each
    __shared__ alignas(32) bf16_t smemP[16 * 32];       // P re-layout   1KB

    // Q tile as A-operand (16 rows x 128 k), kept in registers
    const bf16_t* __restrict__ Qrow = Q + (size_t)(b * cL + q0 + nn) * cD + h * cHD;
    v16bf qa[4];
#pragma unroll
    for (int t = 0; t < 4; ++t) {
        v8bf a0 = *(const v8bf*)(Qrow + t * 32 + hi * 16);
        v8bf a1 = *(const v8bf*)(Qrow + t * 32 + hi * 16 + 8);
#pragma unroll
        for (int i = 0; i < 8; ++i) { qa[t][i] = a0[i]; qa[t][8 + i] = a1[i]; }
    }

    const bf16_t* __restrict__ Kbase = K + (size_t)(b * cL) * cKVD + kh * cHD;
    const bf16_t* __restrict__ Vbase = Vt + (size_t)(b * cKVH + kh) * cHD * cL;

    // per-lane LDS element assignments for the async staging
    const int krow = hi;            // + 2*it  -> rows 0..31
    const int kcol = nn * 8;        // 16B chunk within 256B K row
    const int vrow = lane >> 2;     // + 8*it  -> rows 0..127
    const int vcol = (lane & 3) * 8;// 16B chunk within 64B V row

    // issue one 32x128 K tile + 128x32 V tile (32 async instrs = 16KB)
    auto issue_block = [&](int j0, int buf) {
        const unsigned kofs = (unsigned)(size_t)&smemK[buf][0];
        const unsigned vofs = (unsigned)(size_t)&smemV[buf][0];
#pragma unroll
        for (int it = 0; it < 16; ++it) {
            const int r = it * 2 + krow;
            async_copy16(kofs + (unsigned)(r * 128 + kcol) * 2,
                         Kbase + (size_t)(j0 + r) * cKVD + kcol);
        }
#pragma unroll
        for (int it = 0; it < 16; ++it) {
            const int r = it * 8 + vrow;
            async_copy16(vofs + (unsigned)(r * 32 + vcol) * 2,
                         Vbase + (size_t)r * cL + j0 + vcol);
        }
    };

    v8f acc[8] = {};
    float mrow[8], lrow[8];
#pragma unroll
    for (int c = 0; c < 8; ++c) { mrow[c] = -3.0e38f; lrow[c] = 0.0f; }

    const int jend    = q0 + 15;
    const int nblocks = jend / 32 + 1;

    issue_block(0, 0);                                  // prologue

    for (int blk = 0; blk < nblocks; ++blk) {
        const int j0  = blk * 32;
        const int buf = blk & 1;
        if (blk + 1 < nblocks) {                        // pipeline next block
            issue_block(j0 + 32, buf ^ 1);
            wait_async_32();                            // current block resident
        } else {
            wait_async_0();
        }
        const bf16_t* kt = &smemK[buf][0];
        const bf16_t* vtile = &smemV[buf][0];

        // ---- scores: two 16x16 tiles covering keys j0..j0+31
        v8f s[2] = {};
#pragma unroll
        for (int half = 0; half < 2; ++half) {
#pragma unroll
            for (int t = 0; t < 4; ++t) {
                v16bf kb = *(const v16bf*)(kt + (half * 16 + nn) * 128 + t * 32 + hi * 16);
                s[half] = WMMA_BF16(qa[t], kb, s[half]);
            }
        }

        // ---- scale + causal mask (only diagonal blocks need masking)
        const int jj0 = j0 + nn, jj1 = j0 + 16 + nn;
        const bool need_mask = (j0 + 31 > q0);
#pragma unroll
        for (int c = 0; c < 8; ++c) {
            const int qq = q0 + c + 8 * hi;
            float v0 = s[0][c] * scale, v1 = s[1][c] * scale;
            if (need_mask) {
                if (jj0 > qq) v0 = -1.0e30f;
                if (jj1 > qq) v1 = -1.0e30f;
            }
            s[0][c] = v0; s[1][c] = v1;
        }

        // ---- online softmax (row = per-component, per half-wave; width-16 xor)
#pragma unroll
        for (int c = 0; c < 8; ++c) {
            float mx = fmaxf(s[0][c], s[1][c]);
#pragma unroll
            for (int off = 8; off >= 1; off >>= 1)
                mx = fmaxf(mx, __shfl_xor(mx, off, 16));
            const float mnew  = fmaxf(mrow[c], mx);
            const float alpha = __expf(mrow[c] - mnew);
            mrow[c] = mnew;
            const float p0 = __expf(s[0][c] - mnew);
            const float p1 = __expf(s[1][c] - mnew);
            s[0][c] = p0; s[1][c] = p1;
            float rs = p0 + p1;
#pragma unroll
            for (int off = 8; off >= 1; off >>= 1)
                rs += __shfl_xor(rs, off, 16);
            lrow[c] = lrow[c] * alpha + rs;
#pragma unroll
            for (int t = 0; t < 8; ++t) acc[t][c] *= alpha;
        }

        // ---- restage P (D-layout) -> A-operand layout via LDS
#pragma unroll
        for (int c = 0; c < 8; ++c) {
            smemP[(c + 8 * hi) * 32 + nn]      = (bf16_t)s[0][c];
            smemP[(c + 8 * hi) * 32 + 16 + nn] = (bf16_t)s[1][c];
        }
        __syncthreads();
        v16bf pa;
        {
            v8bf p0 = *(const v8bf*)(smemP + nn * 32 + hi * 16);
            v8bf p1 = *(const v8bf*)(smemP + nn * 32 + hi * 16 + 8);
#pragma unroll
            for (int i = 0; i < 8; ++i) { pa[i] = p0[i]; pa[8 + i] = p1[i]; }
        }
        __syncthreads();

        // ---- O[16x128] += P[16x32] @ V[32x128]  (V^T tile: keys contiguous)
#pragma unroll
        for (int t = 0; t < 8; ++t) {
            v16bf vb = *(const v16bf*)(vtile + (t * 16 + nn) * 32 + hi * 16);
            acc[t] = WMMA_BF16(pa, vb, acc[t]);
        }
    }

    // ---- finalize: divide rows by l, write bf16
#pragma unroll
    for (int c = 0; c < 8; ++c) {
        const float inv = 1.0f / lrow[c];
        const size_t rowoff = (size_t)(b * cL + q0 + c + 8 * hi) * cD + h * cHD;
#pragma unroll
        for (int t = 0; t < 8; ++t)
            O[rowoff + t * 16 + nn] = (bf16_t)(acc[t][c] * inv);
    }
}

// ---------------------------------------------------------------------------
extern "C" void kernel_launch(void* const* d_in, const int* in_sizes, int n_in,
                              void* d_out, int out_size, void* d_ws, size_t ws_size,
                              hipStream_t stream) {
    (void)in_sizes; (void)n_in; (void)out_size; (void)ws_size;
    const float* x  = (const float*)d_in[0];
    const float* Wq = (const float*)d_in[1];
    const float* Wk = (const float*)d_in[2];
    const float* Wv = (const float*)d_in[3];
    const float* Wo = (const float*)d_in[4];
    float* out = (float*)d_out;

    const int M = cB * cL;                       // 4096 tokens
    char* ws = (char*)d_ws;
    size_t off = 0;
    auto carve = [&](size_t bytes) -> char* {
        char* p = ws + off;
        off += (bytes + 255) & ~(size_t)255;
        return p;
    };
    bf16_t* xb    = (bf16_t*)carve((size_t)M * cD * 2);          // 16 MB
    bf16_t* wqb   = (bf16_t*)carve((size_t)cD * cD * 2);         //  8 MB
    bf16_t* wkb   = (bf16_t*)carve((size_t)cKVD * cD * 2);       //  2 MB
    bf16_t* wvb   = (bf16_t*)carve((size_t)cKVD * cD * 2);       //  2 MB
    bf16_t* wob   = (bf16_t*)carve((size_t)cD * cD * 2);         //  8 MB
    bf16_t* qb    = (bf16_t*)carve((size_t)M * cD * 2);          // 16 MB
    bf16_t* kb    = (bf16_t*)carve((size_t)M * cKVD * 2);        //  4 MB
    bf16_t* vb    = (bf16_t*)carve((size_t)M * cKVD * 2);        //  4 MB
    bf16_t* vt    = (bf16_t*)carve((size_t)M * cKVD * 2);        //  4 MB
    bf16_t* attnb = (bf16_t*)carve((size_t)M * cD * 2);          // 16 MB

    auto cast = [&](const float* src, bf16_t* dst, size_t n) {
        cast_f32_bf16<<<2048, 256, 0, stream>>>(src, dst, n);
    };
    cast(x,  xb,  (size_t)M * cD);
    cast(Wq, wqb, (size_t)cD * cD);
    cast(Wk, wkb, (size_t)cKVD * cD);
    cast(Wv, wvb, (size_t)cKVD * cD);
    cast(Wo, wob, (size_t)cD * cD);

    // projections: y = x @ W.T  (bf16 out)
    gemm_xwT<false><<<dim3(M / 16, cD / 256),   128, 0, stream>>>(xb, wqb, nullptr, qb, M, cD,   cD);
    gemm_xwT<false><<<dim3(M / 16, cKVD / 256), 128, 0, stream>>>(xb, wkb, nullptr, kb, M, cKVD, cD);
    gemm_xwT<false><<<dim3(M / 16, cKVD / 256), 128, 0, stream>>>(xb, wvb, nullptr, vb, M, cKVD, cD);

    rope_kernel<<<dim3(cL, cB), 256, 0, stream>>>(qb, kb);
    transpose_v<<<2048, 256, 0, stream>>>(vb, vt);

    gqa_flash_attn<<<dim3(cL / 16, cH, cB), 32, 0, stream>>>(qb, kb, vt, attnb);

    // out = attn @ Wo.T  (f32 out)
    gemm_xwT<true><<<dim3(M / 16, cD / 256), 128, 0, stream>>>(attnb, wob, out, nullptr, M, cD, cD);
}
